// GVPGraphEncoder_5592047420122
// MI455X (gfx1250) — matrix operand order repaired and checked by reading, not compiled, not run on hardware
//
#include <hip/hip_runtime.h>
#include <hip/hip_bf16.h>

// MI455X (gfx1250) GVP-GNN forward. wave32, WMMA f16->f32.
// One wave per 16-row tile; channel-mixing GEMMs via V_WMMA_F32_16X16X32_F16
// with pre-packed per-lane B fragments (2x global_load_b128/lane/frag) and
// b128 LDS A-fragment loads; 3D-vector equivariant ops, norms, gates in f32.

typedef __attribute__((ext_vector_type(16))) _Float16 v16h;
typedef __attribute__((ext_vector_type(8)))  _Float16 v8h;
typedef __attribute__((ext_vector_type(4)))  _Float16 v4h;
typedef __attribute__((ext_vector_type(8)))  float    v8f;

#define N_NODES 50000
#define N_EDGES 400000
#define EPSF 1e-8f

struct LayerW {
  const float* m_b[3];  const float* m_wh[3]; const float* m_wv[3];
  const _Float16* m_ws[3];   // packed B fragments
  const float* f_b[2];  const float* f_wh[2]; const float* f_wv[2];
  const _Float16* f_ws[2];   // packed B fragments
  const float* n0_g; const float* n0_b; const float* n1_g; const float* n1_b;
};

__device__ __forceinline__ float sigmoidf_(float x) { return 1.f / (1.f + __expf(-x)); }

// ---- WMMA fragment helpers (layouts per CDNA5 ISA 7.12.2, wave32) ----
// A 16x32 f16 from LDS: lane m=lane&15, hi=lane>>4; K in {8hi+0..7, 16+8hi+0..7}.
// lda multiple of 16 halves and k0 multiple of 32 -> both groups 16B aligned.
__device__ __forceinline__ v16h ldA_lds(const _Float16* As, int lda, int k0, int lane) {
  const int m = lane & 15, hi = lane >> 4;
  const _Float16* p = As + m * lda + k0 + 8 * hi;
  const v8h lo = *(const v8h*)(p);
  const v8h hk = *(const v8h*)(p + 16);
  return __builtin_shufflevector(lo, hk, 0, 1, 2, 3, 4, 5, 6, 7, 8, 9, 10, 11, 12, 13, 14, 15);
}
// B fragments pre-packed: tile (kt,nt) -> 512 contiguous halves, lane*16 each.
__device__ __forceinline__ v16h ldB_packed(const _Float16* Bp, int ntiles, int kt, int nt, int lane) {
  const _Float16* p = Bp + (((size_t)(kt * ntiles + nt)) << 9) + (lane << 4);
  const v8h lo = *(const v8h*)(p);
  const v8h hk = *(const v8h*)(p + 8);
  return __builtin_shufflevector(lo, hk, 0, 1, 2, 3, 4, 5, 6, 7, 8, 9, 10, 11, 12, 13, 14, 15);
}
// K-outer register-blocked GEMM: NT 16x16 output tiles held in registers,
// A fragment loaded once per K step, B streamed from packed global (L2-resident).
template <int NT>
__device__ __forceinline__ void wmma_block(const _Float16* As, int lda, int Kpad,
                                           const _Float16* Bp, int ntiles, int nt0,
                                           const float* bias, v8f* c, int lane) {
  const int n = lane & 15;
#pragma unroll
  for (int t = 0; t < NT; ++t) {
    const float bv = bias[(nt0 + t) * 16 + n];
#pragma unroll
    for (int r = 0; r < 8; ++r) c[t][r] = bv;
  }
  for (int k0 = 0; k0 < Kpad; k0 += 32) {
    const v16h a = ldA_lds(As, lda, k0, lane);
#pragma unroll
    for (int t = 0; t < NT; ++t) {
      const v16h b = ldB_packed(Bp, ntiles, k0 >> 5, nt0 + t, lane);
      c[t] = __builtin_amdgcn_wmma_f32_16x16x32_f16(false, a, false, b, (short)0, c[t], false, false);
    }
  }
}

// ---------------- small utility kernels ----------------
__global__ void fill_zero_kernel(float* __restrict__ p, int n) {
  const int i = blockIdx.x * 256 + threadIdx.x;
  if (i < n) p[i] = 0.f;
}

__global__ void count_kernel(const int* __restrict__ eidx, float* __restrict__ cnt) {
  const int e = blockIdx.x * 256 + threadIdx.x;
  if (e < N_EDGES) atomicAdd(&cnt[eidx[N_EDGES + e]], 1.f);
}

// f32 (K x N) -> f16 fragments packed per (kt, nt, lane, j); zero-pad K -> Kpad.
__global__ void cvt_pack_kernel(const float* __restrict__ src, _Float16* __restrict__ dst,
                                int K, int N, int Kpad) {
  const int i = blockIdx.x * 256 + threadIdx.x;
  const int ntiles = N >> 4;
  const int total = (Kpad >> 5) * ntiles * 512;
  if (i >= total) return;
  const int j = i & 15;
  const int lane = (i >> 4) & 31;
  const int tile = i >> 9;
  const int kt = tile / ntiles, nt = tile - kt * ntiles;
  const int n = lane & 15, hi = lane >> 4;
  const int Kidx = kt * 32 + 8 * hi + (j & 7) + 16 * (j >> 3);
  const int Nidx = nt * 16 + n;
  dst[i] = (Kidx < K) ? (_Float16)src[Kidx * N + Nidx] : (_Float16)0.f;
}

// ---------------- input GVP: (6,3) -> (128,16), h=16 ----------------
__global__ __launch_bounds__(128) void input_gvp_kernel(
    const float* __restrict__ x_s, const float* __restrict__ x_v,
    const float* __restrict__ bia, const float* __restrict__ wh,
    const float* __restrict__ ws, const float* __restrict__ wv,
    float* __restrict__ hs, float* __restrict__ hv) {
  const int n = blockIdx.x, t = threadIdx.x;
  __shared__ float xv[9], xs[6], vh[48], vn[16];
  if (t < 9) xv[t] = x_v[(size_t)n * 9 + t];
  if (t < 6) xs[t] = x_s[(size_t)n * 6 + t];
  __syncthreads();
  if (t < 48) {
    const int h = t / 3, c = t % 3;
    float a = 0.f;
    for (int i = 0; i < 3; ++i) a += xv[i * 3 + c] * wh[i * 16 + h];
    vh[t] = a;
  }
  __syncthreads();
  if (t < 16) {
    const float* p = &vh[t * 3];
    vn[t] = sqrtf(fmaxf(p[0] * p[0] + p[1] * p[1] + p[2] * p[2], EPSF));
  }
  __syncthreads();
  float a = bia[t];
  for (int i = 0; i < 6; ++i)  a += xs[i] * ws[i * 128 + t];
  for (int h = 0; h < 16; ++h) a += vn[h] * ws[(6 + h) * 128 + t];
  hs[(size_t)n * 128 + t] = fmaxf(a, 0.f);
  if (t < 16) {
    float x = 0.f, y = 0.f, z = 0.f;
    for (int h = 0; h < 16; ++h) {
      const float w = wv[h * 16 + t];
      x += vh[h * 3] * w; y += vh[h * 3 + 1] * w; z += vh[h * 3 + 2] * w;
    }
    const float g = sigmoidf_(sqrtf(fmaxf(x * x + y * y + z * z, EPSF)));
    hv[(size_t)n * 48 + t * 3]     = x * g;
    hv[(size_t)n * 48 + t * 3 + 1] = y * g;
    hv[(size_t)n * 48 + t * 3 + 2] = z * g;
  }
}

// ---------------- edge message kernel: 3 fused GVPs + atomic scatter ----------------
__global__ __launch_bounds__(32) void edge_msg_kernel(
    const float* __restrict__ hs, const float* __restrict__ hv,
    const int* __restrict__ eidx,
    const float* __restrict__ edg_s, const float* __restrict__ edg_v,
    float* __restrict__ agg_s, float* __restrict__ agg_v, LayerW W) {
  const int lane = threadIdx.x;
  const int e0 = blockIdx.x << 4;

  __shared__ __align__(16) _Float16 Abig[16 * 352];
  __shared__ __align__(16) _Float16 Asml[16 * 160];
  __shared__ float vin[16 * 33 * 3];
  __shared__ float vvh[16 * 33 * 3];
  __shared__ float vcur[16 * 16 * 3];
  __shared__ float souts[16 * 128];
  __shared__ int ssrc[16], sdst[16];

  if (lane < 16) {
    ssrc[lane] = eidx[e0 + lane];
    sdst[lane] = eidx[N_EDGES + e0 + lane];
  }
  __syncthreads();

  // GVP0 vector inputs: [v_src(16) | edge_v(1) | v_dst(16)]
  for (int idx = lane; idx < 16 * 33 * 3; idx += 32) {
    const int e = idx / 99, rem = idx % 99, r = rem / 3, c = rem % 3;
    float val;
    if (r < 16)       val = hv[(size_t)ssrc[e] * 48 + r * 3 + c];
    else if (r == 16) val = edg_v[(size_t)(e0 + e) * 3 + c];
    else              val = hv[(size_t)sdst[e] * 48 + (r - 17) * 3 + c];
    vin[idx] = val;
  }
  __syncthreads();
  // vh0 = vin @ wh0 (33x33)
  for (int idx = lane; idx < 16 * 33 * 3; idx += 32) {
    const int e = idx / 99, rem = idx % 99, h = rem / 3, c = rem % 3;
    float acc = 0.f;
    const float* w = W.m_wh[0];
    for (int i = 0; i < 33; ++i) acc += vin[e * 99 + i * 3 + c] * w[i * 33 + h];
    vvh[idx] = acc;
  }
  __syncthreads();
  // A0 = [s_src(128)|edge_s(32)|s_dst(128)|vn0(33)|pad(31)] f16, b128 loads + b64 stores
#pragma unroll 4
  for (int e = 0; e < 16; ++e) {
    {
      const float4 v = *(const float4*)(hs + (size_t)ssrc[e] * 128 + lane * 4);
      v4h t; t[0] = (_Float16)v.x; t[1] = (_Float16)v.y; t[2] = (_Float16)v.z; t[3] = (_Float16)v.w;
      *(v4h*)(&Abig[e * 352 + lane * 4]) = t;
    }
    Abig[e * 352 + 128 + lane] = (_Float16)edg_s[(size_t)(e0 + e) * 32 + lane];
    {
      const float4 v = *(const float4*)(hs + (size_t)sdst[e] * 128 + lane * 4);
      v4h t; t[0] = (_Float16)v.x; t[1] = (_Float16)v.y; t[2] = (_Float16)v.z; t[3] = (_Float16)v.w;
      *(v4h*)(&Abig[e * 352 + 160 + lane * 4]) = t;
    }
  }
  for (int idx = lane; idx < 16 * 64; idx += 32) {
    const int e = idx >> 6, k = idx & 63;
    float val = 0.f;
    if (k < 33) {
      const float* p = &vvh[e * 99 + k * 3];
      val = sqrtf(fmaxf(p[0] * p[0] + p[1] * p[1] + p[2] * p[2], EPSF));
    }
    Abig[e * 352 + 288 + k] = (_Float16)val;
  }
  __syncthreads();
  // s1 = relu(A0 @ ws0 + b0) -> Asml[:,0:128]
  {
    v8f c[8];
    wmma_block<8>(Abig, 352, 352, W.m_ws[0], 8, 0, W.m_b[0], c, lane);
    const int n = lane & 15, hi = lane >> 4;
#pragma unroll
    for (int t = 0; t < 8; ++t)
#pragma unroll
      for (int r = 0; r < 8; ++r)
        Asml[(r + 8 * hi) * 160 + t * 16 + n] = (_Float16)fmaxf(c[t][r], 0.f);
  }
  // v1 = gate(vh0 @ wv0 (33x16))
  for (int p = lane; p < 16 * 16; p += 32) {
    const int e = p >> 4, o = p & 15;
    float x = 0.f, y = 0.f, z = 0.f;
    const float* w = W.m_wv[0];
    for (int h = 0; h < 33; ++h) {
      const float wv = w[h * 16 + o];
      x += vvh[e * 99 + h * 3] * wv; y += vvh[e * 99 + h * 3 + 1] * wv; z += vvh[e * 99 + h * 3 + 2] * wv;
    }
    const float g = sigmoidf_(sqrtf(fmaxf(x * x + y * y + z * z, EPSF)));
    vcur[e * 48 + o * 3] = x * g; vcur[e * 48 + o * 3 + 1] = y * g; vcur[e * 48 + o * 3 + 2] = z * g;
  }
  __syncthreads();
  // ---- GVP1 ---- vh1 = v1 @ wh1 (16x16)
  for (int idx = lane; idx < 16 * 16 * 3; idx += 32) {
    const int e = idx / 48, rem = idx % 48, h = rem / 3, c = rem % 3;
    float acc = 0.f;
    const float* w = W.m_wh[1];
    for (int i = 0; i < 16; ++i) acc += vcur[e * 48 + i * 3 + c] * w[i * 16 + h];
    vvh[e * 48 + h * 3 + c] = acc;
  }
  __syncthreads();
  for (int idx = lane; idx < 16 * 32; idx += 32) {
    const int e = idx >> 5, k = idx & 31;
    float val = 0.f;
    if (k < 16) {
      const float* p = &vvh[e * 48 + k * 3];
      val = sqrtf(fmaxf(p[0] * p[0] + p[1] * p[1] + p[2] * p[2], EPSF));
    }
    Asml[e * 160 + 128 + k] = (_Float16)val;
  }
  __syncthreads();
  {
    v8f c[8];
    wmma_block<8>(Asml, 160, 160, W.m_ws[1], 8, 0, W.m_b[1], c, lane);
    const int n = lane & 15, hi = lane >> 4;
#pragma unroll
    for (int t = 0; t < 8; ++t)
#pragma unroll
      for (int r = 0; r < 8; ++r)
        Abig[(r + 8 * hi) * 160 + t * 16 + n] = (_Float16)fmaxf(c[t][r], 0.f);
  }
  for (int p = lane; p < 16 * 16; p += 32) {
    const int e = p >> 4, o = p & 15;
    float x = 0.f, y = 0.f, z = 0.f;
    const float* w = W.m_wv[1];
    for (int h = 0; h < 16; ++h) {
      const float wv = w[h * 16 + o];
      x += vvh[e * 48 + h * 3] * wv; y += vvh[e * 48 + h * 3 + 1] * wv; z += vvh[e * 48 + h * 3 + 2] * wv;
    }
    const float g = sigmoidf_(sqrtf(fmaxf(x * x + y * y + z * z, EPSF)));
    vcur[e * 48 + o * 3] = x * g; vcur[e * 48 + o * 3 + 1] = y * g; vcur[e * 48 + o * 3 + 2] = z * g;
  }
  __syncthreads();
  // ---- GVP2 (no activations) ----
  for (int idx = lane; idx < 16 * 16 * 3; idx += 32) {
    const int e = idx / 48, rem = idx % 48, h = rem / 3, c = rem % 3;
    float acc = 0.f;
    const float* w = W.m_wh[2];
    for (int i = 0; i < 16; ++i) acc += vcur[e * 48 + i * 3 + c] * w[i * 16 + h];
    vvh[e * 48 + h * 3 + c] = acc;
  }
  __syncthreads();
  for (int idx = lane; idx < 16 * 32; idx += 32) {
    const int e = idx >> 5, k = idx & 31;
    float val = 0.f;
    if (k < 16) {
      const float* p = &vvh[e * 48 + k * 3];
      val = sqrtf(fmaxf(p[0] * p[0] + p[1] * p[1] + p[2] * p[2], EPSF));
    }
    Abig[e * 160 + 128 + k] = (_Float16)val;
  }
  __syncthreads();
  {
    v8f c[8];
    wmma_block<8>(Abig, 160, 160, W.m_ws[2], 8, 0, W.m_b[2], c, lane);
    const int n = lane & 15, hi = lane >> 4;
#pragma unroll
    for (int t = 0; t < 8; ++t)
#pragma unroll
      for (int r = 0; r < 8; ++r)
        souts[(r + 8 * hi) * 128 + t * 16 + n] = c[t][r];
  }
  for (int p = lane; p < 16 * 16; p += 32) {
    const int e = p >> 4, o = p & 15;
    float x = 0.f, y = 0.f, z = 0.f;
    const float* w = W.m_wv[2];
    for (int h = 0; h < 16; ++h) {
      const float wv = w[h * 16 + o];
      x += vvh[e * 48 + h * 3] * wv; y += vvh[e * 48 + h * 3 + 1] * wv; z += vvh[e * 48 + h * 3 + 2] * wv;
    }
    vcur[e * 48 + o * 3] = x; vcur[e * 48 + o * 3 + 1] = y; vcur[e * 48 + o * 3 + 2] = z;
  }
  __syncthreads();
  // scatter-add into destination node accumulators
  for (int idx = lane; idx < 16 * 128; idx += 32) {
    const int e = idx >> 7;
    atomicAdd(&agg_s[(size_t)sdst[e] * 128 + (idx & 127)], souts[idx]);
  }
  for (int idx = lane; idx < 16 * 48; idx += 32) {
    const int e = idx / 48;
    atomicAdd(&agg_v[(size_t)sdst[e] * 48 + (idx % 48)], vcur[idx]);
  }
}

// ---------------- node update: residual + norm0 + FF(2 GVPs) + norm1 ----------------
__global__ __launch_bounds__(32) void node_update_kernel(
    float* __restrict__ hs, float* __restrict__ hv,
    const float* __restrict__ agg_s, const float* __restrict__ agg_v,
    const float* __restrict__ cnt, LayerW W) {
  const int lane = threadIdx.x;
  const int n0 = blockIdx.x << 4;

  __shared__ __align__(16) _Float16 A2[16 * 544];
  __shared__ __align__(16) _Float16 A1[16 * 160];
  __shared__ float vin[16 * 32 * 3];
  __shared__ float vvh[16 * 32 * 3];
  __shared__ __align__(16) float sres[16 * 128];
  __shared__ float vres[16 * 48];
  __shared__ float souts[16 * 128];
  __shared__ float mu_[16], rs_[16], vs_[16], rcn[16];

  if (lane < 16) rcn[lane] = 1.f / fmaxf(cnt[n0 + lane], 1.f);
  __syncthreads();
#pragma unroll 4
  for (int e = 0; e < 16; ++e) {
    const size_t gb = (size_t)(n0 + e) * 128 + lane * 4;
    const float4 a = *(const float4*)(hs + gb);
    const float4 b = *(const float4*)(agg_s + gb);
    const float rc = rcn[e];
    float4 o;
    o.x = a.x + b.x * rc; o.y = a.y + b.y * rc; o.z = a.z + b.z * rc; o.w = a.w + b.w * rc;
    *(float4*)(&sres[e * 128 + lane * 4]) = o;
  }
  for (int idx = lane; idx < 16 * 48; idx += 32) {
    const int m = idx / 48, j = idx % 48;
    vres[idx] = hv[(size_t)(n0 + m) * 48 + j] + agg_v[(size_t)(n0 + m) * 48 + j] * rcn[m];
  }
  __syncthreads();
  if (lane < 16) {
    float mu = 0.f;
    for (int k = 0; k < 128; ++k) mu += sres[lane * 128 + k];
    mu *= (1.f / 128.f);
    float var = 0.f;
    for (int k = 0; k < 128; ++k) { const float d = sres[lane * 128 + k] - mu; var += d * d; }
    mu_[lane] = mu; rs_[lane] = rsqrtf(var * (1.f / 128.f) + 1e-5f);
    float vs = 0.f;
    for (int i = 0; i < 16; ++i) {
      const float* p = &vres[lane * 48 + i * 3];
      vs += fmaxf(p[0] * p[0] + p[1] * p[1] + p[2] * p[2], EPSF);
    }
    vs_[lane] = rsqrtf(vs * (1.f / 16.f));
  }
  __syncthreads();
  for (int idx = lane; idx < 16 * 128; idx += 32) {
    const int m = idx >> 7, k = idx & 127;
    sres[idx] = (sres[idx] - mu_[m]) * rs_[m] * W.n0_g[k] + W.n0_b[k];
  }
  for (int idx = lane; idx < 16 * 48; idx += 32) vres[idx] *= vs_[idx / 48];
  __syncthreads();
  // ff0: (128,16)->(512,32), h=32
  for (int idx = lane; idx < 16 * 32 * 3; idx += 32) {
    const int e = idx / 96, rem = idx % 96, h = rem / 3, c = rem % 3;
    float acc = 0.f;
    const float* w = W.f_wh[0];
    for (int i = 0; i < 16; ++i) acc += vres[e * 48 + i * 3 + c] * w[i * 32 + h];
    vvh[idx] = acc;
  }
  __syncthreads();
#pragma unroll 4
  for (int e = 0; e < 16; ++e) {
    const float4 v = *(const float4*)(&sres[e * 128 + lane * 4]);
    v4h t; t[0] = (_Float16)v.x; t[1] = (_Float16)v.y; t[2] = (_Float16)v.z; t[3] = (_Float16)v.w;
    *(v4h*)(&A1[e * 160 + lane * 4]) = t;
  }
  for (int idx = lane; idx < 16 * 32; idx += 32) {
    const int e = idx >> 5, h = idx & 31;
    const float* p = &vvh[e * 96 + h * 3];
    A1[e * 160 + 128 + h] = (_Float16)sqrtf(fmaxf(p[0] * p[0] + p[1] * p[1] + p[2] * p[2], EPSF));
  }
  __syncthreads();
  // s_h = relu(A1 @ f_ws0 + f_b0) -> A2[:,0:512], 4 register-blocked groups of 8 tiles
#pragma unroll
  for (int g = 0; g < 4; ++g) {
    v8f c[8];
    wmma_block<8>(A1, 160, 160, W.f_ws[0], 32, g * 8, W.f_b[0], c, lane);
    const int n = lane & 15, hi = lane >> 4;
#pragma unroll
    for (int t = 0; t < 8; ++t)
#pragma unroll
      for (int r = 0; r < 8; ++r)
        A2[(r + 8 * hi) * 544 + (g * 8 + t) * 16 + n] = (_Float16)fmaxf(c[t][r], 0.f);
  }
  for (int p = lane; p < 16 * 32; p += 32) {
    const int e = p >> 5, o = p & 31;
    float x = 0.f, y = 0.f, z = 0.f;
    const float* w = W.f_wv[0];
    for (int h = 0; h < 32; ++h) {
      const float wv = w[h * 32 + o];
      x += vvh[e * 96 + h * 3] * wv; y += vvh[e * 96 + h * 3 + 1] * wv; z += vvh[e * 96 + h * 3 + 2] * wv;
    }
    const float g = sigmoidf_(sqrtf(fmaxf(x * x + y * y + z * z, EPSF)));
    vin[e * 96 + o * 3] = x * g; vin[e * 96 + o * 3 + 1] = y * g; vin[e * 96 + o * 3 + 2] = z * g;
  }
  __syncthreads();
  // ff1: (512,32)->(128,16), h=32, no acts
  for (int idx = lane; idx < 16 * 32 * 3; idx += 32) {
    const int e = idx / 96, rem = idx % 96, h = rem / 3, c = rem % 3;
    float acc = 0.f;
    const float* w = W.f_wh[1];
    for (int i = 0; i < 32; ++i) acc += vin[e * 96 + i * 3 + c] * w[i * 32 + h];
    vvh[idx] = acc;
  }
  __syncthreads();
  for (int idx = lane; idx < 16 * 32; idx += 32) {
    const int e = idx >> 5, h = idx & 31;
    const float* p = &vvh[e * 96 + h * 3];
    A2[e * 544 + 512 + h] = (_Float16)sqrtf(fmaxf(p[0] * p[0] + p[1] * p[1] + p[2] * p[2], EPSF));
  }
  __syncthreads();
  {
    v8f c[8];
    wmma_block<8>(A2, 544, 544, W.f_ws[1], 8, 0, W.f_b[1], c, lane);
    const int n = lane & 15, hi = lane >> 4;
#pragma unroll
    for (int t = 0; t < 8; ++t)
#pragma unroll
      for (int r = 0; r < 8; ++r) {
        const int m = r + 8 * hi;
        souts[m * 128 + t * 16 + n] = sres[m * 128 + t * 16 + n] + c[t][r];
      }
  }
  for (int p = lane; p < 16 * 16; p += 32) {
    const int e = p >> 4, o = p & 15;
    float x = 0.f, y = 0.f, z = 0.f;
    const float* w = W.f_wv[1];
    for (int h = 0; h < 32; ++h) {
      const float wv = w[h * 16 + o];
      x += vvh[e * 96 + h * 3] * wv; y += vvh[e * 96 + h * 3 + 1] * wv; z += vvh[e * 96 + h * 3 + 2] * wv;
    }
    vres[e * 48 + o * 3] += x; vres[e * 48 + o * 3 + 1] += y; vres[e * 48 + o * 3 + 2] += z;
  }
  __syncthreads();
  if (lane < 16) {
    float mu = 0.f;
    for (int k = 0; k < 128; ++k) mu += souts[lane * 128 + k];
    mu *= (1.f / 128.f);
    float var = 0.f;
    for (int k = 0; k < 128; ++k) { const float d = souts[lane * 128 + k] - mu; var += d * d; }
    mu_[lane] = mu; rs_[lane] = rsqrtf(var * (1.f / 128.f) + 1e-5f);
    float vs = 0.f;
    for (int i = 0; i < 16; ++i) {
      const float* p = &vres[lane * 48 + i * 3];
      vs += fmaxf(p[0] * p[0] + p[1] * p[1] + p[2] * p[2], EPSF);
    }
    vs_[lane] = rsqrtf(vs * (1.f / 16.f));
  }
  __syncthreads();
  for (int idx = lane; idx < 16 * 128; idx += 32) {
    const int m = idx >> 7, k = idx & 127;
    hs[(size_t)(n0 + m) * 128 + k] = (souts[idx] - mu_[m]) * rs_[m] * W.n1_g[k] + W.n1_b[k];
  }
  for (int idx = lane; idx < 16 * 48; idx += 32) {
    hv[(size_t)(n0 + idx / 48) * 48 + idx % 48] = vres[idx] * vs_[idx / 48];
  }
}

// ---------------- final layernorm ----------------
__global__ __launch_bounds__(128) void final_norm_kernel(
    float* __restrict__ hs, float* __restrict__ hv,
    const float* __restrict__ g, const float* __restrict__ b) {
  const int n = blockIdx.x, t = threadIdx.x;
  __shared__ float red[128];
  __shared__ float mu_s, rs_s, vs_s;
  const float x = hs[(size_t)n * 128 + t];
  red[t] = x; __syncthreads();
#pragma unroll
  for (int s = 64; s > 0; s >>= 1) { if (t < s) red[t] += red[t + s]; __syncthreads(); }
  if (t == 0) mu_s = red[0] * (1.f / 128.f);
  __syncthreads();
  const float d = x - mu_s;
  red[t] = d * d; __syncthreads();
#pragma unroll
  for (int s = 64; s > 0; s >>= 1) { if (t < s) red[t] += red[t + s]; __syncthreads(); }
  if (t == 0) rs_s = rsqrtf(red[0] * (1.f / 128.f) + 1e-5f);
  __syncthreads();
  hs[(size_t)n * 128 + t] = d * rs_s * g[t] + b[t];
  if (t < 16) {
    const float* p = &hv[(size_t)n * 48 + t * 3];
    red[t] = fmaxf(p[0] * p[0] + p[1] * p[1] + p[2] * p[2], EPSF);
  }
  __syncthreads();
  if (t == 0) {
    float s = 0.f;
    for (int i = 0; i < 16; ++i) s += red[i];
    vs_s = rsqrtf(s * (1.f / 16.f));
  }
  __syncthreads();
  if (t < 48) hv[(size_t)n * 48 + t] *= vs_s;
}

// ---------------- host launcher ----------------
extern "C" void kernel_launch(void* const* d_in, const int* in_sizes, int n_in,
                              void* d_out, int out_size, void* d_ws, size_t ws_size,
                              hipStream_t stream) {
  (void)in_sizes; (void)n_in; (void)out_size; (void)ws_size;
  const float* x_s   = (const float*)d_in[0];
  const float* x_v   = (const float*)d_in[1];
  const int*   eidx  = (const int*)d_in[2];
  const float* edg_s = (const float*)d_in[3];
  const float* edg_v = (const float*)d_in[4];
  // params flattened in JAX pytree order (sorted dict keys):
  // 5..8: inp {b, wh, ws, wv}; per layer (base=9+24*l): ff0{b,wh,ws,wv}, ff1{...},
  // msg0{...}, msg1{...}, msg2{...}, norm0{b,g}, norm1{b,g}; 81: norm.b, 82: norm.g

  float* hs = (float*)d_out;
  float* hv = hs + (size_t)N_NODES * 128;

  char* base = (char*)d_ws;
  size_t off = 0;
  auto take = [&](size_t bytes) -> void* {
    void* p = base + off;
    off = (off + bytes + 255) & ~(size_t)255;
    return p;
  };
  float* agg_s = (float*)take((size_t)N_NODES * 128 * 4);
  float* agg_v = (float*)take((size_t)N_NODES * 48 * 4);
  float* cnt   = (float*)take((size_t)N_NODES * 4);
  _Float16 *mws0[3], *mws1[3], *mws2[3], *fws0[3], *fws1[3];
  for (int l = 0; l < 3; ++l) {
    mws0[l] = (_Float16*)take((size_t)352 * 128 * 2);
    mws1[l] = (_Float16*)take((size_t)160 * 128 * 2);
    mws2[l] = (_Float16*)take((size_t)160 * 128 * 2);
    fws0[l] = (_Float16*)take((size_t)160 * 512 * 2);
    fws1[l] = (_Float16*)take((size_t)544 * 128 * 2);
  }

  // degree counts
  fill_zero_kernel<<<(N_NODES + 255) / 256, 256, 0, stream>>>(cnt, N_NODES);
  count_kernel<<<(N_EDGES + 255) / 256, 256, 0, stream>>>(eidx, cnt);

  // fp32 -> fp16 fragment-packed weight conversion
  auto cvt = [&](const void* src, _Float16* dst, int K, int N, int Kpad) {
    const int tot = Kpad * N;
    cvt_pack_kernel<<<(tot + 255) / 256, 256, 0, stream>>>((const float*)src, dst, K, N, Kpad);
  };
  for (int l = 0; l < 3; ++l) {
    const int b9 = 9 + 24 * l;
    cvt(d_in[b9 + 10], mws0[l], 321, 128, 352);
    cvt(d_in[b9 + 14], mws1[l], 144, 128, 160);
    cvt(d_in[b9 + 18], mws2[l], 144, 128, 160);
    cvt(d_in[b9 + 2],  fws0[l], 160, 512, 160);
    cvt(d_in[b9 + 6],  fws1[l], 544, 128, 544);
  }

  // input GVP
  input_gvp_kernel<<<N_NODES, 128, 0, stream>>>(
      x_s, x_v, (const float*)d_in[5], (const float*)d_in[6],
      (const float*)d_in[7], (const float*)d_in[8], hs, hv);

  for (int l = 0; l < 3; ++l) {
    const int b9 = 9 + 24 * l;
    LayerW W;
    for (int g = 0; g < 3; ++g) {
      W.m_b[g]  = (const float*)d_in[b9 + 8 + 4 * g + 0];
      W.m_wh[g] = (const float*)d_in[b9 + 8 + 4 * g + 1];
      W.m_wv[g] = (const float*)d_in[b9 + 8 + 4 * g + 3];
    }
    W.m_ws[0] = mws0[l]; W.m_ws[1] = mws1[l]; W.m_ws[2] = mws2[l];
    for (int g = 0; g < 2; ++g) {
      W.f_b[g]  = (const float*)d_in[b9 + 4 * g + 0];
      W.f_wh[g] = (const float*)d_in[b9 + 4 * g + 1];
      W.f_wv[g] = (const float*)d_in[b9 + 4 * g + 3];
    }
    W.f_ws[0] = fws0[l]; W.f_ws[1] = fws1[l];
    W.n0_b = (const float*)d_in[b9 + 20]; W.n0_g = (const float*)d_in[b9 + 21];
    W.n1_b = (const float*)d_in[b9 + 22]; W.n1_g = (const float*)d_in[b9 + 23];

    fill_zero_kernel<<<(N_NODES * 128 + 255) / 256, 256, 0, stream>>>(agg_s, N_NODES * 128);
    fill_zero_kernel<<<(N_NODES * 48 + 255) / 256, 256, 0, stream>>>(agg_v, N_NODES * 48);
    edge_msg_kernel<<<N_EDGES / 16, 32, 0, stream>>>(hs, hv, eidx, edg_s, edg_v, agg_s, agg_v, W);
    node_update_kernel<<<N_NODES / 16, 32, 0, stream>>>(hs, hv, agg_s, agg_v, cnt, W);
  }

  final_norm_kernel<<<N_NODES, 128, 0, stream>>>(
      hs, hv, (const float*)d_in[82], (const float*)d_in[81]);
}
